// MultiHeadedAttention_29205777613423
// MI455X (gfx1250) — matrix operand order repaired and checked by reading, not compiled, not run on hardware
//
#include <hip/hip_runtime.h>
#include <math.h>

typedef __attribute__((ext_vector_type(16))) _Float16 v16h;
typedef __attribute__((ext_vector_type(8)))  _Float16 v8h;
typedef __attribute__((ext_vector_type(8)))  float    v8f;

#define DEV static __device__ __forceinline__

static constexpr int B_ = 4, C_ = 256, HW_ = 4096;
static constexpr size_t PER_BATCH_QKV = 1048576; // halves per batch in Q/K/Vt

// ---------------- WMMA fragment helpers (wave32, 16x16x32 f16) ----------------
// A fragment: row = lane%16; halves h -> K = kk + (lane/16)*16 + h
DEV v16h frag_a(const _Float16* M, int row, int ld, int kk, int lane) {
  const _Float16* r = M + (size_t)row * ld + kk + ((lane >> 4) << 4);
  v8h lo = *(const v8h*)r;
  v8h hi = *(const v8h*)(r + 8);
  v16h f;
#pragma unroll
  for (int i = 0; i < 8; ++i) { f[i] = lo[i]; f[i + 8] = hi[i]; }
  return f;
}

// B fragment (B[k][n]): col n = colBase + lane%16; halves h -> K = kk + (lane/16)*16 + h
// T stored transposed [n][k] row-major (k contiguous), ld in halves.
DEV v16h frag_b(const _Float16* T, int colBase, int ld, int kk, int lane) {
  const _Float16* r = T + (size_t)(colBase + (lane & 15)) * ld + kk + ((lane >> 4) << 4);
  return *(const v16h*)r;
}

DEV v8f wmma(v16h a, v16h b, v8f c) {
  return __builtin_amdgcn_wmma_f32_16x16x32_f16(false, a, false, b, (short)0, c, false, false);
}
// C/D element r of v8f: row = r + (lane/16)*8, col = lane%16

// ---------------- packing kernels ----------------
__global__ void k_zero(float* s) { s[threadIdx.x] = 0.f; }

__global__ void k_pack_x(const float* __restrict__ x, _Float16* __restrict__ xt) {
  int i = blockIdx.x * 256 + threadIdx.x;          // [b][p][c], c fastest
  int b = i >> 20;
  int rem = i & 1048575;
  int p = rem >> 8;
  int c = rem & 255;
  xt[i] = (_Float16)x[(size_t)b * 1048576 + (size_t)c * 4096 + p];
}

__global__ void k_pack_w(const float* __restrict__ wq, const float* __restrict__ wk,
                         const float* __restrict__ wv, const float* __restrict__ wo,
                         _Float16* __restrict__ wh, _Float16* __restrict__ wc) {
  int i = blockIdx.x * 256 + threadIdx.x;
  if (i < 196608) {                                 // wq|wk|wv, [o][c] f16
    const float* src = (i < 65536) ? wq : (i < 131072 ? wk : wv);
    wh[i] = (_Float16)src[i & 65535];
  } else {                                          // conv w: [o][(ky*3+kx)*256 + c]
    int j = i - 196608;
    int o = j / 2304;
    int t = j % 2304;
    int kyx = t >> 8;
    int c = t & 255;
    wc[j] = (_Float16)wo[(size_t)(o * 256 + c) * 9 + kyx];
  }
}

// ---------------- QKV projection: Out[o,p] = sum_c W[o,c] * X[c,p] + b[o] ----------------
// 1 o-tile x 2 p-tiles per wave: each A fragment (weights) feeds 2 WMMAs.
// Writes Q,K token-major [b][s][n][d], V feature-major [b][s][d][n] (f16).
__global__ __launch_bounds__(256) void k_qkv(
    const _Float16* __restrict__ xt, const _Float16* __restrict__ wh,
    const float* __restrict__ bq, const float* __restrict__ bk, const float* __restrict__ bv,
    _Float16* __restrict__ Qh, _Float16* __restrict__ Kh, _Float16* __restrict__ Vt) {
  int wid = blockIdx.x * 8 + (threadIdx.x >> 5);
  int lane = threadIdx.x & 31;
  int b = wid >> 11;               // 16 otiles * 128 ptile-pairs per batch
  int ot = (wid >> 7) & 15;
  int pt = wid & 127;              // 32-wide spatial tile
  const _Float16* xb = xt + (size_t)b * HW_ * C_;
  const _Float16* wqh = wh;
  const _Float16* wkh = wh + 65536;
  const _Float16* wvh = wh + 131072;
  v8f cq[2] = {}, ck[2] = {}, cv[2] = {};
  int orow = ot * 16 + (lane & 15);
  for (int kk = 0; kk < 256; kk += 32) {
    v16h bx0 = frag_b(xb, pt * 32,      256, kk, lane);
    v16h bx1 = frag_b(xb, pt * 32 + 16, 256, kk, lane);
    v16h aq = frag_a(wqh, orow, 256, kk, lane);
    cq[0] = wmma(aq, bx0, cq[0]);  cq[1] = wmma(aq, bx1, cq[1]);
    v16h ak = frag_a(wkh, orow, 256, kk, lane);
    ck[0] = wmma(ak, bx0, ck[0]);  ck[1] = wmma(ak, bx1, ck[1]);
    v16h av = frag_a(wvh, orow, 256, kk, lane);
    cv[0] = wmma(av, bx0, cv[0]);  cv[1] = wmma(av, bx1, cv[1]);
  }
  size_t bbase = (size_t)b * PER_BATCH_QKV;
#pragma unroll
  for (int pi = 0; pi < 2; ++pi) {
#pragma unroll
    for (int r = 0; r < 8; ++r) {
      int o = ot * 16 + r + ((lane >> 4) << 3);
      int p = pt * 32 + pi * 16 + (lane & 15);
      int s = o >> 6, cl = o & 63;
      int lg = 3 - s;                       // log2(patch)
      int ph = 1 << lg;
      int h = p >> 6, w = p & 63;
      int oh = h >> lg, py = h & (ph - 1);
      int ow = w >> lg, px = w & (ph - 1);
      int outw = 64 >> lg;
      int n = oh * outw + ow;
      int Ds = 64 << (2 * lg);
      int Ns = 4096 >> (2 * lg);
      int dfeat = (cl << (2 * lg)) + (py << lg) + px;
      size_t base = bbase + (size_t)s * 262144;
      Qh[base + (size_t)n * Ds + dfeat] = (_Float16)(cq[pi][r] + bq[o]);
      Kh[base + (size_t)n * Ds + dfeat] = (_Float16)(ck[pi][r] + bk[o]);
      Vt[base + (size_t)dfeat * Ns + n] = (_Float16)(cv[pi][r] + bv[o]);
    }
  }
}

// ---------------- scores (scales 0-2): S = Q K^T / sqrt(D), 2x2 register tiling ----------------
template <int DD, int NN>
__global__ __launch_bounds__(256) void k_scores(const _Float16* __restrict__ Q,
                                                const _Float16* __restrict__ K,
                                                float* __restrict__ S, int sOff) {
  int wid = blockIdx.x * 8 + (threadIdx.x >> 5);
  int lane = threadIdx.x & 31;
  constexpr int NT2 = NN / 32;
  int b = wid / (NT2 * NT2);
  int rem = wid % (NT2 * NT2);
  int mt2 = rem / NT2, nt2 = rem % NT2;
  const _Float16* Qb = Q + (size_t)b * PER_BATCH_QKV + sOff;
  const _Float16* Kb = K + (size_t)b * PER_BATCH_QKV + sOff;
  v8f acc[2][2] = {};
  for (int kk = 0; kk < DD; kk += 32) {
    v16h qa0 = frag_a(Qb, mt2 * 32 +      (lane & 15), DD, kk, lane);
    v16h qa1 = frag_a(Qb, mt2 * 32 + 16 + (lane & 15), DD, kk, lane);
    v16h kb0 = frag_b(Kb, nt2 * 32,      DD, kk, lane);
    v16h kb1 = frag_b(Kb, nt2 * 32 + 16, DD, kk, lane);
    acc[0][0] = wmma(qa0, kb0, acc[0][0]);
    acc[0][1] = wmma(qa0, kb1, acc[0][1]);
    acc[1][0] = wmma(qa1, kb0, acc[1][0]);
    acc[1][1] = wmma(qa1, kb1, acc[1][1]);
  }
  const float scl = 1.0f / sqrtf((float)DD);
  float* Sb = S + (size_t)b * NN * NN;
#pragma unroll
  for (int qi = 0; qi < 2; ++qi)
#pragma unroll
    for (int ni = 0; ni < 2; ++ni)
#pragma unroll
      for (int r = 0; r < 8; ++r) {
        int m = mt2 * 32 + qi * 16 + r + ((lane >> 4) << 3);
        int n = nt2 * 32 + ni * 16 + (lane & 15);
        Sb[(size_t)m * NN + n] = acc[qi][ni][r] * scl;
      }
}

// ---------------- softmax row-wise, write f16 P ----------------
template <int NN>
__global__ __launch_bounds__(256) void k_softmax(const float* __restrict__ S,
                                                 _Float16* __restrict__ P) {
  __shared__ float red[256];
  int row = blockIdx.x;
  int tid = threadIdx.x;
  const float* Sr = S + (size_t)row * NN;
  float mx = -1e30f;
  for (int i = tid; i < NN; i += 256) mx = fmaxf(mx, Sr[i]);
  red[tid] = mx; __syncthreads();
  for (int s = 128; s > 0; s >>= 1) { if (tid < s) red[tid] = fmaxf(red[tid], red[tid + s]); __syncthreads(); }
  mx = red[0]; __syncthreads();
  float sm = 0.f;
  for (int i = tid; i < NN; i += 256) sm += expf(Sr[i] - mx);
  red[tid] = sm; __syncthreads();
  for (int s = 128; s > 0; s >>= 1) { if (tid < s) red[tid] += red[tid + s]; __syncthreads(); }
  float inv = 1.0f / red[0];
  _Float16* Pr = P + (size_t)row * NN;
  for (int i = tid; i < NN; i += 256) Pr[i] = (_Float16)(expf(Sr[i] - mx) * inv);
}

// ---------------- PV (scales 0-2): Y = P V, 2x2 tiling, scatter channel-last ----------------
template <int DD, int NN, int SC>
__global__ __launch_bounds__(256) void k_pv(const _Float16* __restrict__ P,
                                            const _Float16* __restrict__ Vt,
                                            _Float16* __restrict__ y, int sOff) {
  int wid = blockIdx.x * 8 + (threadIdx.x >> 5);
  int lane = threadIdx.x & 31;
  constexpr int QT2 = NN / 32, DT2 = DD / 32;
  int b = wid / (QT2 * DT2);
  int rem = wid % (QT2 * DT2);
  int qt2 = rem / DT2, dt2 = rem % DT2;
  const _Float16* Pb = P + (size_t)b * NN * NN;
  const _Float16* Vb = Vt + (size_t)b * PER_BATCH_QKV + sOff;
  v8f acc[2][2] = {};
  for (int mm = 0; mm < NN; mm += 32) {
    v16h pa0 = frag_a(Pb, qt2 * 32 +      (lane & 15), NN, mm, lane);
    v16h pa1 = frag_a(Pb, qt2 * 32 + 16 + (lane & 15), NN, mm, lane);
    v16h vb0 = frag_b(Vb, dt2 * 32,      NN, mm, lane);
    v16h vb1 = frag_b(Vb, dt2 * 32 + 16, NN, mm, lane);
    acc[0][0] = wmma(pa0, vb0, acc[0][0]);
    acc[0][1] = wmma(pa0, vb1, acc[0][1]);
    acc[1][0] = wmma(pa1, vb0, acc[1][0]);
    acc[1][1] = wmma(pa1, vb1, acc[1][1]);
  }
  constexpr int LG = 3 - SC;
  constexpr int PH = 1 << LG;
  constexpr int OUTW = 64 >> LG;
#pragma unroll
  for (int qi = 0; qi < 2; ++qi)
#pragma unroll
    for (int di = 0; di < 2; ++di)
#pragma unroll
      for (int r = 0; r < 8; ++r) {
        int n = qt2 * 32 + qi * 16 + r + ((lane >> 4) << 3);
        int d = dt2 * 32 + di * 16 + (lane & 15);
        int oh = n / OUTW, ow = n % OUTW;
        int cl = d >> (2 * LG);
        int rr = d & ((1 << (2 * LG)) - 1);
        int py = rr >> LG, px = rr & (PH - 1);
        int h = oh * PH + py, w = ow * PH + px;
        int c = SC * 64 + cl;
        y[(((size_t)b * 64 + h) * 64 + w) * 256 + c] = (_Float16)acc[qi][di][r];
      }
}

// ---------------- flash attention, scale 3: N=4096, D=64 ----------------
// 8 waves/block share one K/V stream staged through LDS (8x L2-traffic cut);
// next tile prefetched with global_prefetch.
__global__ __launch_bounds__(256) void k_flash(const _Float16* __restrict__ Q,
                                               const _Float16* __restrict__ K,
                                               const _Float16* __restrict__ Vt,
                                               _Float16* __restrict__ y) {
  __shared__ __align__(32) _Float16 ldsK[32 * 64];   // [key][d]
  __shared__ __align__(32) _Float16 ldsV[64 * 32];   // [d][key]
  __shared__ __align__(32) _Float16 ldsP[8][512];    // per-wave 16x32 P transpose
  int tid = threadIdx.x;
  int lane = tid & 31;
  int wslot = tid >> 5;
  int b = blockIdx.x >> 5;                 // 32 blocks per batch
  int qt = (blockIdx.x & 31) * 8 + wslot;
  const size_t sOff = 3 * 262144;
  const _Float16* Qb = Q + (size_t)b * PER_BATCH_QKV + sOff;
  const _Float16* Kb = K + (size_t)b * PER_BATCH_QKV + sOff;
  const _Float16* Vb = Vt + (size_t)b * PER_BATCH_QKV + sOff;
  int qrow = qt * 16 + (lane & 15);
  v16h aq0 = frag_a(Qb, qrow, 64, 0, lane);
  v16h aq1 = frag_a(Qb, qrow, 64, 32, lane);
  v8f o0 = {}, o1 = {}, o2 = {}, o3 = {};
  float mrow[8], lrow[8];
#pragma unroll
  for (int r = 0; r < 8; ++r) { mrow[r] = -1e30f; lrow[r] = 0.f; }
  const float scl = 0.125f;                // 1/sqrt(64)
  _Float16* slab = ldsP[wslot];
  // staging indices (one 16B chunk of K and one of V per thread per step)
  int kkey = tid >> 3, kd8 = tid & 7;      // K: 32 keys x 8 chunks
  int vd = tid >> 2, vk8 = tid & 3;        // V: 64 d    x 4 chunks
  for (int kt = 0; kt < 128; ++kt) {       // 32 keys per step
    __syncthreads();                        // previous tile fully consumed
    *(uint4*)&ldsK[kkey * 64 + kd8 * 8] =
        *(const uint4*)&Kb[(size_t)(kt * 32 + kkey) * 64 + kd8 * 8];
    *(uint4*)&ldsV[vd * 32 + vk8 * 8] =
        *(const uint4*)&Vb[(size_t)vd * 4096 + kt * 32 + vk8 * 8];
    if (kt + 1 < 128) {
      __builtin_prefetch(&Kb[(size_t)((kt + 1) * 32 + kkey) * 64 + kd8 * 8], 0, 0);
      __builtin_prefetch(&Vb[(size_t)vd * 4096 + (kt + 1) * 32 + vk8 * 8], 0, 0);
    }
    __syncthreads();                        // tile visible to all waves
    v8f sA = {}, sB = {};
    sA = wmma(aq0, frag_b(ldsK, 0,  64, 0,  lane), sA);
    sB = wmma(aq0, frag_b(ldsK, 16, 64, 0,  lane), sB);
    sA = wmma(aq1, frag_b(ldsK, 0,  64, 32, lane), sA);
    sB = wmma(aq1, frag_b(ldsK, 16, 64, 32, lane), sB);
    float pa[8], pb[8];
#pragma unroll
    for (int r = 0; r < 8; ++r) {
      float x0 = sA[r] * scl, x1 = sB[r] * scl;
      float t = fmaxf(x0, x1);
#pragma unroll
      for (int m = 1; m < 16; m <<= 1) t = fmaxf(t, __shfl_xor(t, m, 32));
      float mn = fmaxf(mrow[r], t);
      float corr = __expf(mrow[r] - mn);
      float e0 = __expf(x0 - mn), e1 = __expf(x1 - mn);
      float srow = e0 + e1;
#pragma unroll
      for (int m = 1; m < 16; m <<= 1) srow += __shfl_xor(srow, m, 32);
      lrow[r] = lrow[r] * corr + srow;
      mrow[r] = mn;
      o0[r] *= corr; o1[r] *= corr; o2[r] *= corr; o3[r] *= corr;
      pa[r] = e0; pb[r] = e1;
    }
    // C-layout -> A-layout transpose through wave-private LDS slab (DS in-order)
#pragma unroll
    for (int r = 0; r < 8; ++r) {
      int row = r + ((lane >> 4) << 3);
      int col = lane & 15;
      slab[row * 32 + col]      = (_Float16)pa[r];
      slab[row * 32 + 16 + col] = (_Float16)pb[r];
    }
    v16h ap = frag_a(slab, lane & 15, 32, 0, lane);
    o0 = wmma(ap, frag_b(ldsV, 0,  32, 0, lane), o0);
    o1 = wmma(ap, frag_b(ldsV, 16, 32, 0, lane), o1);
    o2 = wmma(ap, frag_b(ldsV, 32, 32, 0, lane), o2);
    o3 = wmma(ap, frag_b(ldsV, 48, 32, 0, lane), o3);
  }
#pragma unroll
  for (int r = 0; r < 8; ++r) {
    int n = qt * 16 + r + ((lane >> 4) << 3);
    float inv = 1.0f / lrow[r];
    int h = n >> 6, w = n & 63;
    size_t base = (((size_t)b * 64 + h) * 64 + w) * 256 + 192;
    int dloc = lane & 15;
    y[base + 0  + dloc] = (_Float16)(o0[r] * inv);
    y[base + 16 + dloc] = (_Float16)(o1[r] * inv);
    y[base + 32 + dloc] = (_Float16)(o2[r] * inv);
    y[base + 48 + dloc] = (_Float16)(o3[r] * inv);
  }
}

// ---------------- 3x3 conv (implicit GEMM, K=9*256) + bias + BN stats ----------------
// 512-thread block = 16 waves; the block's padded 3x18x256 input neighborhood is
// staged once in LDS (zero padding baked in) and shared by all 16 o-tiles.
__global__ __launch_bounds__(512) void k_conv(const _Float16* __restrict__ y,
                                              const _Float16* __restrict__ wc,
                                              const float* __restrict__ b_out,
                                              float* __restrict__ z,
                                              float* __restrict__ stats) {
  __shared__ __align__(32) _Float16 ldsY[3 * 18 * 256];   // 27 KB
  int tid = threadIdx.x;
  int lane = tid & 31;
  int wv = tid >> 5;                       // o-tile index 0..15
  int b = blockIdx.x >> 8;                 // 256 p-tiles per batch
  int pt = blockIdx.x & 255;
  int h0 = pt >> 2;
  int w0 = (pt & 3) << 4;
  const _Float16* yb = y + (size_t)b * HW_ * 256;
  for (int ch = tid; ch < 1728; ch += 512) {       // 16B chunks
    int c8 = ch & 31;
    int cx = (ch >> 5) % 18;
    int ry = (ch >> 5) / 18;
    int hh = h0 - 1 + ry;
    int ww = w0 - 1 + cx;
    uint4 val = {0u, 0u, 0u, 0u};
    if ((unsigned)hh < 64u && (unsigned)ww < 64u)
      val = *(const uint4*)&yb[((size_t)hh * 64 + ww) * 256 + c8 * 8];
    *(uint4*)&ldsY[(ry * 18 + cx) * 256 + c8 * 8] = val;
  }
  __syncthreads();
  v8f acc = {};
  int orow = wv * 16 + (lane & 15);
  for (int ky = 0; ky < 3; ++ky) {
    for (int kx = 0; kx < 3; ++kx) {
      const _Float16* bbase =
          &ldsY[((ky * 18) + (lane & 15) + kx) * 256 + ((lane >> 4) << 4)];
      int kbase = (ky * 3 + kx) * 256;
      for (int cc = 0; cc < 256; cc += 32) {
        v16h a = frag_a(wc, orow, 2304, kbase + cc, lane);
        v16h bb = *(const v16h*)(bbase + cc);
        acc = wmma(a, bb, acc);
      }
    }
  }
  int p = pt * 16 + (lane & 15);
  float* zb = z + (size_t)b * 256 * HW_;
#pragma unroll
  for (int r = 0; r < 8; ++r) {
    int o = wv * 16 + r + ((lane >> 4) << 3);
    float val = acc[r] + b_out[o];
    zb[(size_t)o * HW_ + p] = val;
    float s1 = val, s2 = val * val;
#pragma unroll
    for (int m = 1; m < 16; m <<= 1) { s1 += __shfl_xor(s1, m, 32); s2 += __shfl_xor(s2, m, 32); }
    if ((lane & 15) == 0) {
      atomicAdd(&stats[o], s1);
      atomicAdd(&stats[256 + o], s2);
    }
  }
}

// ---------------- BN (training stats) + LeakyReLU(0.2) ----------------
__global__ void k_bn(const float* __restrict__ z, const float* __restrict__ stats,
                     const float* __restrict__ gamma, const float* __restrict__ beta,
                     float* __restrict__ out) {
  int i = blockIdx.x * 256 + threadIdx.x;
  int o = (i >> 12) & 255;
  const float n = 16384.f;
  float mean = stats[o] / n;
  float var = stats[256 + o] / n - mean * mean;
  float inv = rsqrtf(var + 1e-5f);
  float v = (z[i] - mean) * inv * gamma[o] + beta[o];
  out[i] = v > 0.f ? v : 0.2f * v;
}

// ---------------- host launch ----------------
extern "C" void kernel_launch(void* const* d_in, const int* in_sizes, int n_in,
                              void* d_out, int out_size, void* d_ws, size_t ws_size,
                              hipStream_t stream) {
  (void)in_sizes; (void)n_in; (void)out_size; (void)ws_size;
  const float* x     = (const float*)d_in[0];
  const float* wq    = (const float*)d_in[1];
  const float* bq    = (const float*)d_in[2];
  const float* wk    = (const float*)d_in[3];
  const float* bk    = (const float*)d_in[4];
  const float* wv    = (const float*)d_in[5];
  const float* bv    = (const float*)d_in[6];
  const float* w_out = (const float*)d_in[7];
  const float* b_out = (const float*)d_in[8];
  const float* gamma = (const float*)d_in[9];
  const float* beta  = (const float*)d_in[10];

  char* ws = (char*)d_ws;
  constexpr size_t O_XT = 0;                         // 8 MB  f16 x, [b][p][c]
  constexpr size_t O_WH = O_XT + 8388608;            // 384 KB f16 wq|wk|wv
  constexpr size_t O_WC = O_WH + 393216;             // 1.125 MB f16 conv w packed
  constexpr size_t O_Q  = O_WC + 1179648;            // 8 MB
  constexpr size_t O_K  = O_Q + 8388608;             // 8 MB
  constexpr size_t O_VT = O_K + 8388608;             // 8 MB
  constexpr size_t O_S  = O_VT + 8388608;            // 17.06 MB f32 scores (scales 0-2)
  constexpr size_t O_P  = O_S + 17891328;            // 8.53 MB f16 P
  constexpr size_t O_Y  = O_P + 8945664;             // 8 MB f16 y channel-last
  constexpr size_t O_Z  = O_Y + 8388608;             // 16 MB f32 conv out
  constexpr size_t O_ST = O_Z + 16777216;            // 2 KB stats

  _Float16* xt = (_Float16*)(ws + O_XT);
  _Float16* wh = (_Float16*)(ws + O_WH);
  _Float16* wc = (_Float16*)(ws + O_WC);
  _Float16* Qh = (_Float16*)(ws + O_Q);
  _Float16* Kh = (_Float16*)(ws + O_K);
  _Float16* Vt = (_Float16*)(ws + O_VT);
  float*    S0 = (float*)(ws + O_S);
  float*    S1 = S0 + 16384;
  float*    S2 = S1 + 262144;
  _Float16* P0 = (_Float16*)(ws + O_P);
  _Float16* P1 = P0 + 16384;
  _Float16* P2 = P1 + 262144;
  _Float16* y  = (_Float16*)(ws + O_Y);
  float*    z  = (float*)(ws + O_Z);
  float*    st = (float*)(ws + O_ST);

  k_zero<<<1, 512, 0, stream>>>(st);
  k_pack_x<<<16384, 256, 0, stream>>>(x, xt);
  k_pack_w<<<3072, 256, 0, stream>>>(wq, wk, wv, w_out, wh, wc);
  k_qkv<<<1024, 256, 0, stream>>>(xt, wh, bq, bk, bv, Qh, Kh, Vt);

  k_scores<4096, 64><<<2, 256, 0, stream>>>(Qh, Kh, S0, 0);
  k_scores<1024, 256><<<32, 256, 0, stream>>>(Qh, Kh, S1, 262144);
  k_scores<256, 1024><<<512, 256, 0, stream>>>(Qh, Kh, S2, 524288);

  k_softmax<64><<<256, 256, 0, stream>>>(S0, P0);
  k_softmax<256><<<1024, 256, 0, stream>>>(S1, P1);
  k_softmax<1024><<<4096, 256, 0, stream>>>(S2, P2);

  k_pv<4096, 64, 0><<<128, 256, 0, stream>>>(P0, Vt, y, 0);
  k_pv<1024, 256, 1><<<128, 256, 0, stream>>>(P1, Vt, y, 262144);
  k_pv<256, 1024, 2><<<128, 256, 0, stream>>>(P2, Vt, y, 524288);

  k_flash<<<128, 256, 0, stream>>>(Qh, Kh, Vt, y);

  k_conv<<<1024, 512, 0, stream>>>(y, wc, b_out, z, st);
  k_bn<<<16384, 256, 0, stream>>>(z, st, gamma, beta, (float*)d_out);
}
